// MovingWindowEuclideanForecaster_39341900431909
// MI455X (gfx1250) — matrix operand order repaired and checked by reading, not compiled, not run on hardware
//
#include <hip/hip_runtime.h>
#include <math.h>

// ---------------- problem constants ----------------
#define NSEG     14
#define SEGLEN   24
#define DD       64
#define HH       128
#define PREDSEG  8
#define LL       336
#define FF       512
#define BF       32768           // B * F

// ---------------- launch shape ----------------
#define BDIM            64       // 2 waves of 32
#define WAVES           2
#define ROWS_PER_WAVE   16
#define ROWS_PER_BLOCK  32

// ---------------- LDS strides (ushort elems, padded so u32 stride is odd) ----
#define ENC_KP  98               // enc weights  B^T [64][96+2]
#define DW1_KP  130              // dyn_W1  B^T [64][128+2]
#define DW2_KP  66               // dyn_W2  B^T [64][64+2]
#define RW1_KP  66               // rec_W1  B^T [128][64+2]
#define RW2_KP  130              // rec_W2  B^T [32][128+2]  (N padded 24->32)
#define ZSTR    66               // z / vel / h row stride (64+2)
#define ZSLOT   (16 * ZSTR)      // one 16x64 tile slot
#define GSTR    130              // g row stride (128+2)

typedef __attribute__((ext_vector_type(16))) __bf16 bf16x16;
typedef __attribute__((ext_vector_type(2)))  __bf16 bf16x2;
typedef __attribute__((ext_vector_type(8)))  float  f32x8;

union ABFrag { unsigned int u[8]; bf16x16 v; };

// Native bf16 converts: let the backend emit v_cvt_pk_bf16_f32 / scalar cvt
// instead of a 4-op software RNE sequence.
__device__ __forceinline__ unsigned short f2bf(float x) {
  return __builtin_bit_cast(unsigned short, (__bf16)x);
}
__device__ __forceinline__ float bf2f(unsigned short h) {
  return (float)__builtin_bit_cast(__bf16, h);
}
__device__ __forceinline__ float bf2f_lo(unsigned int w) {
  return __uint_as_float(w << 16);
}
__device__ __forceinline__ float bf2f_hi(unsigned int w) {
  return __uint_as_float(w & 0xFFFF0000u);
}
__device__ __forceinline__ unsigned int pack2(float lo, float hi) {
#if __has_builtin(__builtin_amdgcn_cvt_pk_bf16_f32)
  return __builtin_bit_cast(unsigned int, __builtin_amdgcn_cvt_pk_bf16_f32(lo, hi));
#else
  bf16x2 t;
  t[0] = (__bf16)lo;
  t[1] = (__bf16)hi;
  return __builtin_bit_cast(unsigned int, t);
#endif
}

__device__ __forceinline__ float gelu_tanh(float x) {
  float x3 = x * x * x;
  return 0.5f * x * (1.f + tanhf(0.7978845608028654f * (x + 0.044715f * x3)));
}

__device__ __forceinline__ f32x8 wmma_bf16(bf16x16 a, bf16x16 b, f32x8 c) {
  // D = A(16x32 bf16) * B(32x16 bf16) + C(16x16 f32)
  return __builtin_amdgcn_wmma_f32_16x16x32_bf16(false, a, false, b,
                                                 (short)0, c, false, false);
}

// A fragment (16x32 bf16) from row-major [16][stride] LDS buffer, k slice at
// element offset k_off.  ISA 7.12.2 16-bit A layout: lane m = l&15,
// hi-halfwave gets +8 on K; VGPR v holds K pair {2v,2v+1} (v<4) / {16+2(v-4),..}.
__device__ __forceinline__ bf16x16 load_A(const unsigned short* buf, int u32_stride,
                                          int k_off, int lane) {
  const unsigned int* p = (const unsigned int*)buf;
  int m  = lane & 15;
  int kb = (lane & 16) ? 8 : 0;
  ABFrag a;
#pragma unroll
  for (int v = 0; v < 8; ++v) {
    int k0 = k_off + kb + ((v < 4) ? (2 * v) : (16 + 2 * (v - 4)));
    a.u[v] = p[m * u32_stride + (k0 >> 1)];
  }
  return a.v;
}

// B fragment (32x16 bf16) from transposed weights stored [N][K]-padded in LDS.
// lane l holds column n = n_off + (l&15); K = k_off + ((l&16)?8:0) + (j<8 ? j : 8+j).
__device__ __forceinline__ bf16x16 load_Bt(const unsigned short* bt, int u32_stride,
                                           int n_off, int k_off, int lane) {
  const unsigned int* p = (const unsigned int*)bt;
  int n  = n_off + (lane & 15);
  int kb = k_off + ((lane & 16) ? 8 : 0);
  ABFrag b;
#pragma unroll
  for (int j = 0; j < 8; ++j) b.u[j] = p[n * u32_stride + (kb >> 1) + ((j < 4) ? j : (8 + (j - 4)))];
  return b.v;
}

__global__ __launch_bounds__(BDIM) void
forecaster_kernel(const float* __restrict__ trend, const float* __restrict__ seas_c,
                  const float* __restrict__ seas_f, const float* __restrict__ resid,
                  const float* __restrict__ enc_W, const float* __restrict__ enc_b,
                  const float* __restrict__ dyn_W1, const float* __restrict__ dyn_b1,
                  const float* __restrict__ dyn_W2, const float* __restrict__ dyn_b2,
                  const float* __restrict__ step_size,
                  const float* __restrict__ rec_W1, const float* __restrict__ rec_b1,
                  const float* __restrict__ rec_W2, const float* __restrict__ rec_b2,
                  float* __restrict__ out) {
  __shared__ __align__(16) unsigned short ls_encW[64 * ENC_KP];
  __shared__ __align__(16) unsigned short ls_dW1[64 * DW1_KP];
  __shared__ __align__(16) unsigned short ls_dW2[64 * DW2_KP];
  __shared__ __align__(16) unsigned short ls_rW1[128 * RW1_KP];
  __shared__ __align__(16) unsigned short ls_rW2[32 * RW2_KP];
  __shared__ __align__(16) unsigned short ls_z[WAVES * NSEG * ZSLOT];
  __shared__ __align__(16) unsigned short ls_vel[WAVES * ZSLOT];
  __shared__ __align__(16) unsigned short ls_h[WAVES * ZSLOT];
  __shared__ __align__(16) unsigned short ls_g[WAVES * 16 * GSTR];

  const int tid = threadIdx.x;

  // ---- stage all weights to LDS as bf16, transposed [N][K] with padding ----
  for (int i = tid; i < 64 * 96; i += BDIM) {           // enc: K=96 (4 comps x 24)
    int d = i / 96, k = i % 96;
    ls_encW[d * ENC_KP + k] = f2bf(enc_W[k * 64 + d]);
  }
  for (int i = tid; i < 64 * 128; i += BDIM) {          // dyn_W1: (128,64)
    int d = i / 128, k = i % 128;
    ls_dW1[d * DW1_KP + k] = f2bf(dyn_W1[k * 64 + d]);
  }
  for (int i = tid; i < 64 * 64; i += BDIM) {           // dyn_W2: (64,64)
    int d = i / 64, k = i % 64;
    ls_dW2[d * DW2_KP + k] = f2bf(dyn_W2[k * 64 + d]);
  }
  for (int i = tid; i < 128 * 64; i += BDIM) {          // rec_W1: (64,128)
    int n = i / 64, k = i % 64;
    ls_rW1[n * RW1_KP + k] = f2bf(rec_W1[k * 128 + n]);
  }
  for (int i = tid; i < 32 * 128; i += BDIM) {          // rec_W2: (128,24)->[32][128]
    int n = i / 128, k = i % 128;
    ls_rW2[n * RW2_KP + k] =
        (n < SEGLEN) ? f2bf(rec_W2[k * SEGLEN + n]) : (unsigned short)0;
  }
  __syncthreads();

  const int lane = tid & 31;
  const int w    = tid >> 5;
  const int nl   = lane & 15;
  const int hi   = (lane >> 4) & 1;
  const int kb8  = hi ? 8 : 0;

  unsigned short* zW   = ls_z   + w * NSEG * ZSLOT;
  unsigned short* velW = ls_vel + w * ZSLOT;
  unsigned short* hW   = ls_h   + w * ZSLOT;
  unsigned short* gW   = ls_g   + w * 16 * GSTR;

  const int row0 = blockIdx.x * ROWS_PER_BLOCK + w * ROWS_PER_WAVE;
  const int b    = row0 >> 9;              // / FF
  const int f0   = row0 & (FF - 1);
  const size_t bL = (size_t)b * LL;

  // exponential-decay mixing coefficients (telescoped: velocity = sum cf[n]*z[n])
  float wv[13];
  {
    float p = 1.f, tot = 0.f;
    for (int i = 12; i >= 0; --i) { wv[i] = p; tot += p; p *= 0.9f; }
    float inv = 1.f / tot;
    for (int i = 0; i < 13; ++i) wv[i] *= inv;
  }
  float cf[14];
  cf[0] = -wv[0]; cf[13] = wv[12];
  for (int n = 1; n < 13; ++n) cf[n] = wv[n - 1] - wv[n];

  const float sgate = 1.f / (1.f + expf(-step_size[0]));

  // per-lane bias values (C-layout column d = t*16 + nl)
  float encb4[4], db1_4[4], db2_4[4], rb1_8[8], rb2_2[2];
#pragma unroll
  for (int t = 0; t < 4; ++t) {
    int d = t * 16 + nl;
    encb4[t] = enc_b[d] + enc_b[64 + d] + enc_b[128 + d] + enc_b[192 + d];
    db1_4[t] = dyn_b1[d];
    db2_4[t] = dyn_b2[d];
  }
#pragma unroll
  for (int t = 0; t < 8; ++t) rb1_8[t] = rec_b1[t * 16 + nl];
  rb2_2[0] = rec_b2[nl];
  rb2_2[1] = (nl < 8) ? rec_b2[16 + nl] : 0.f;

  const f32x8 zero8 = {0.f, 0.f, 0.f, 0.f, 0.f, 0.f, 0.f, 0.f};

  // ================= encoder: z[n] = sum_c comp_c @ enc_W_c + enc_b ==========
  const int f = f0 + nl;                    // lane's row -> feature column
#pragma unroll 1
  for (int n = 0; n < NSEG; ++n) {
    f32x8 acc[4] = {zero8, zero8, zero8, zero8};
#pragma unroll
    for (int ks = 0; ks < 3; ++ks) {        // K = 96 -> 3 x 32
      ABFrag a;
#pragma unroll
      for (int v = 0; v < 8; ++v) {
        int k0 = ks * 32 + kb8 + ((v < 4) ? (2 * v) : (16 + 2 * (v - 4)));
        int c = k0 / 24, s = k0 % 24;       // component / in-segment index
        const float* base = (c == 0) ? trend : (c == 1) ? seas_c
                                      : (c == 2) ? seas_f : resid;
        const float* q = base + (bL + (size_t)(n * SEGLEN + s)) * FF + f;
        a.u[v] = pack2(q[0], q[FF]);        // (s, s+1) pair, stride F in memory
      }
#pragma unroll
      for (int t = 0; t < 4; ++t)
        acc[t] = wmma_bf16(a.v, load_Bt(ls_encW, ENC_KP / 2, t * 16, ks * 32, lane),
                           acc[t]);
    }
#pragma unroll
    for (int t = 0; t < 4; ++t) {
      int d = t * 16 + nl;
#pragma unroll
      for (int v = 0; v < 8; ++v)
        zW[n * ZSLOT + (v + 8 * hi) * ZSTR + d] = f2bf(acc[t][v] + encb4[t]);
    }
  }

  // ================= recurrent rollout =======================================
  int start = 0;
#pragma unroll 1
  for (int step = 0; step < PREDSEG; ++step) {
    // ---- velocity = sum_n cf[n] * z[n]; lanes each own 16 bf16 PAIRS --------
#pragma unroll 1
    for (int t = 0; t < 16; ++t) {
      int p  = t * 32 + lane;               // pair index (512 pairs = 16x64)
      int mm = p >> 5;                      // 32 pairs per row
      int dp = (p & 31) * 2;                // element column of pair
      float a0 = 0.f, a1 = 0.f;
#pragma unroll
      for (int i = 0; i < NSEG; ++i) {
        int sl = start + i; if (sl >= NSEG) sl -= NSEG;
        unsigned int zz =
            *(const unsigned int*)(zW + sl * ZSLOT + mm * ZSTR + dp);
        a0 += cf[i] * bf2f_lo(zz);
        a1 += cf[i] * bf2f_hi(zz);
      }
      *(unsigned int*)(velW + mm * ZSTR + dp) = pack2(a0, a1);
    }
    int slot_last = start + 13; if (slot_last >= NSEG) slot_last -= NSEG;
    const unsigned short* zlast = zW + slot_last * ZSLOT;

    // ---- h = tanh([z_last | velocity] @ dyn_W1 + b1)   K=128, N=64
    f32x8 h4[4] = {zero8, zero8, zero8, zero8};
#pragma unroll
    for (int ks = 0; ks < 4; ++ks) {
      const unsigned short* ab = (ks < 2) ? zlast : velW;
      bf16x16 a = load_A(ab, ZSTR / 2, (ks & 1) * 32, lane);
#pragma unroll
      for (int t = 0; t < 4; ++t)
        h4[t] = wmma_bf16(a, load_Bt(ls_dW1, DW1_KP / 2, t * 16, ks * 32, lane), h4[t]);
    }
#pragma unroll
    for (int t = 0; t < 4; ++t) {
      int d = t * 16 + nl;
#pragma unroll
      for (int v = 0; v < 8; ++v)
        hW[(v + 8 * hi) * ZSTR + d] = f2bf(tanhf(h4[t][v] + db1_4[t]));
    }

    // ---- zdelta = h @ dyn_W2     K=64, N=64
    f32x8 z4[4] = {zero8, zero8, zero8, zero8};
#pragma unroll
    for (int ks = 0; ks < 2; ++ks) {
      bf16x16 a = load_A(hW, ZSTR / 2, ks * 32, lane);
#pragma unroll
      for (int t = 0; t < 4; ++t)
        z4[t] = wmma_bf16(a, load_Bt(ls_dW2, DW2_KP / 2, t * 16, ks * 32, lane), z4[t]);
    }

    // ---- z_next = z_last + s * (velocity + zdelta + b2); overwrite oldest slot
    int slot_new = start;
    unsigned short* znew = zW + slot_new * ZSLOT;
#pragma unroll
    for (int t = 0; t < 4; ++t) {
      int d = t * 16 + nl;
#pragma unroll
      for (int v = 0; v < 8; ++v) {
        int mm = v + 8 * hi;
        float zl = bf2f(zlast[mm * ZSTR + d]);
        float ve = bf2f(velW[mm * ZSTR + d]);
        float zraw = zl + ve + z4[t][v] + db2_4[t];
        znew[mm * ZSTR + d] = f2bf(zl + sgate * (zraw - zl));
      }
    }

    // ---- g = gelu(z_next @ rec_W1 + rb1)   K=64, N=128
    f32x8 g8[8] = {zero8, zero8, zero8, zero8, zero8, zero8, zero8, zero8};
#pragma unroll
    for (int ks = 0; ks < 2; ++ks) {
      bf16x16 a = load_A(znew, ZSTR / 2, ks * 32, lane);
#pragma unroll
      for (int t = 0; t < 8; ++t)
        g8[t] = wmma_bf16(a, load_Bt(ls_rW1, RW1_KP / 2, t * 16, ks * 32, lane), g8[t]);
    }
#pragma unroll
    for (int t = 0; t < 8; ++t) {
      int d = t * 16 + nl;
#pragma unroll
      for (int v = 0; v < 8; ++v)
        gW[(v + 8 * hi) * GSTR + d] = f2bf(gelu_tanh(g8[t][v] + rb1_8[t]));
    }

    // ---- pred = g @ rec_W2 + rb2    K=128, N=24 (padded to 32)
    f32x8 p2[2] = {zero8, zero8};
#pragma unroll
    for (int ks = 0; ks < 4; ++ks) {
      bf16x16 a = load_A(gW, GSTR / 2, ks * 32, lane);
#pragma unroll
      for (int t = 0; t < 2; ++t)
        p2[t] = wmma_bf16(a, load_Bt(ls_rW2, RW2_KP / 2, t * 16, ks * 32, lane), p2[t]);
    }
#pragma unroll
    for (int t = 0; t < 2; ++t) {
      int col = t * 16 + nl;
      if (col < SEGLEN) {
#pragma unroll
        for (int v = 0; v < 8; ++v) {
          int mm = v + 8 * hi;
          out[((size_t)b * (PREDSEG * SEGLEN) + step * SEGLEN + col) * FF + f0 + mm] =
              p2[t][v] + rb2_2[t];
        }
      }
    }

    ++start; if (start >= NSEG) start = 0;
  }
}

extern "C" void kernel_launch(void* const* d_in, const int* in_sizes, int n_in,
                              void* d_out, int out_size, void* d_ws, size_t ws_size,
                              hipStream_t stream) {
  (void)in_sizes; (void)n_in; (void)out_size; (void)d_ws; (void)ws_size;
  const float* trend = (const float*)d_in[0];
  const float* sc    = (const float*)d_in[1];
  const float* sf    = (const float*)d_in[2];
  const float* res   = (const float*)d_in[3];
  const float* encW  = (const float*)d_in[4];
  const float* encb  = (const float*)d_in[5];
  const float* dW1   = (const float*)d_in[6];
  const float* db1   = (const float*)d_in[7];
  const float* dW2   = (const float*)d_in[8];
  const float* db2   = (const float*)d_in[9];
  const float* ss    = (const float*)d_in[10];
  const float* rW1   = (const float*)d_in[11];
  const float* rb1   = (const float*)d_in[12];
  const float* rW2   = (const float*)d_in[13];
  const float* rb2   = (const float*)d_in[14];
  float* out = (float*)d_out;

  dim3 grid(BF / ROWS_PER_BLOCK);   // 1024 blocks of 2 waves (32 rows each)
  dim3 block(BDIM);
  hipLaunchKernelGGL(forecaster_kernel, grid, block, 0, stream,
                     trend, sc, sf, res, encW, encb, dW1, db1, dW2, db2, ss,
                     rW1, rb1, rW2, rb2, out);
}
